// BoundaryAttentionHead_25340307046481
// MI455X (gfx1250) — matrix operand order repaired and checked by reading, not compiled, not run on hardware
//
#include <hip/hip_runtime.h>
#include <hip/hip_bf16.h>

typedef _Float16 v16h __attribute__((ext_vector_type(16)));
typedef _Float16 v8h  __attribute__((ext_vector_type(8)));
typedef _Float16 v4h  __attribute__((ext_vector_type(4)));
typedef float    v8f  __attribute__((ext_vector_type(8)));

#define CCH 128   // in_channels

// ---------------------------------------------------------------------------
// Kernel 1: pack W1 [128,64] f32 row-major into f16 WMMA B-fragments.
// Fragment (nt,kc) covers B = W1[kc*32 .. kc*32+31][nt*16 .. nt*16+15].
// B 32x16 order used here (contiguous k, same as A): lane<16 -> col N=lane, K=0..15 (halves 0..15);
// lane>=16 -> col N=lane-16, K=16..31. Stored contiguously: 16 halves/lane.
// ---------------------------------------------------------------------------
__global__ void pack_w1_kernel(const float* __restrict__ W1,
                               _Float16* __restrict__ pw1) {
    const int tid  = threadIdx.x;        // 0..511
    const int frag = tid >> 5;           // 0..15 == nt*4 + kc
    const int lane = tid & 31;
    const int nt   = frag >> 2;
    const int kc   = frag & 3;
    const int n    = nt * 16 + (lane & 15);
    const int kb   = kc * 32 + ((lane >> 4) & 1) * 16;
    _Float16* dst  = pw1 + (size_t)(frag * 32 + lane) * 16;
#pragma unroll
    for (int j = 0; j < 16; ++j)
        dst[j] = (_Float16)W1[(size_t)(kb + j) * 64 + n];
}

// ---------------------------------------------------------------------------
// Kernel 2: per-node neighborhood variance. One wave per node.
// Lane l owns channels 4l..4l+3 (float4). Neighbor indices broadcast by shfl.
// Writes variance as f16 [N,128] into workspace.
// ---------------------------------------------------------------------------
__global__ void variance_kernel(const float* __restrict__ x,
                                const int* __restrict__ src,
                                _Float16* __restrict__ vws,
                                int N, int K, float invK) {
    const int lane = threadIdx.x & 31;
    const int wid  = threadIdx.x >> 5;
    const int node = blockIdx.x * 8 + wid;      // 256 threads = 8 waves/block
    if (node >= N) return;

    const float4 c = *(const float4*)(x + (size_t)node * CCH + lane * 4);

    // lanes 0..K-1 each load one neighbor index; broadcast inside the loop
    int sv = (lane < K) ? src[(size_t)node * K + lane] : 0;

    float4 acc = make_float4(0.f, 0.f, 0.f, 0.f);
    for (int j = 0; j < K; ++j) {
        const int s = __shfl(sv, j, 32);
        const float4 nb = *(const float4*)(x + (size_t)s * CCH + lane * 4);
        const float dx = nb.x - c.x, dy = nb.y - c.y;
        const float dz = nb.z - c.z, dw = nb.w - c.w;
        acc.x += dx * dx; acc.y += dy * dy;
        acc.z += dz * dz; acc.w += dw * dw;
    }

    v4h o;
    o[0] = (_Float16)(acc.x * invK);
    o[1] = (_Float16)(acc.y * invK);
    o[2] = (_Float16)(acc.z * invK);
    o[3] = (_Float16)(acc.w * invK);
    *(v4h*)(vws + (size_t)node * CCH + lane * 4) = o;
}

// ---------------------------------------------------------------------------
// Kernel 3: fused MLP via v_wmma_f32_16x16x32_f16.
// One wave per 16-node tile: acc[nt] (nt=0..3) = var_tile @ W1[:,nt*16..].
// A 16x32 f16 order used here (contiguous k, same as B): lane<16 row M=lane, K kb+{0..15};
// lane>=16 same row M=lane-16 with K kb+16+{0..15}.
// D 16x16 f32 layout: lane = col (N=lane&15), VGPR r = row r + 8*(lane>>4).
// ---------------------------------------------------------------------------
__global__ void mlp_kernel(const _Float16* __restrict__ vws,
                           const _Float16* __restrict__ pw1,
                           const float* __restrict__ b1,
                           const float* __restrict__ W2,
                           const float* __restrict__ b2,
                           float* __restrict__ out,
                           int N, int nTiles) {
    const int lane = threadIdx.x & 31;
    const int wid  = threadIdx.x >> 5;
    const int tile = blockIdx.x * 8 + wid;
    if (tile >= nTiles) return;                 // wave-uniform: EXEC stays full

    const int ln = lane & 15;
    const int hi = lane >> 4;

    v8f acc[4];
#pragma unroll
    for (int nt = 0; nt < 4; ++nt)
#pragma unroll
        for (int r = 0; r < 8; ++r) acc[nt][r] = 0.f;

#pragma unroll
    for (int kc = 0; kc < 4; ++kc) {
        const _Float16* arow =
            vws + (size_t)(tile * 16 + ln) * CCH + kc * 32 + hi * 16;
        const v8h lo = *(const v8h*)arow;
        const v8h hh = *(const v8h*)(arow + 8);
        v16h a;
#pragma unroll
        for (int j = 0; j < 8; ++j) { a[j] = lo[j]; a[j + 8] = hh[j]; }

#pragma unroll
        for (int nt = 0; nt < 4; ++nt) {
            const v16h b =
                *(const v16h*)(pw1 + (size_t)((nt * 4 + kc) * 32 + lane) * 16);
            acc[nt] = __builtin_amdgcn_wmma_f32_16x16x32_f16(
                false, a, false, b, (short)0, acc[nt], false, false);
        }
    }

    // h = relu(acc + b1), partial out = h * W2 ; lane covers cols nt*16+ln
    float s[8];
#pragma unroll
    for (int r = 0; r < 8; ++r) s[r] = 0.f;
#pragma unroll
    for (int nt = 0; nt < 4; ++nt) {
        const float bb = b1[nt * 16 + ln];
        const float ww = W2[nt * 16 + ln];
#pragma unroll
        for (int r = 0; r < 8; ++r) {
            float h = acc[nt][r] + bb;
            h = h > 0.f ? h : 0.f;
            s[r] += h * ww;
        }
    }

    // reduce over the 16 lanes of each half (xor masks < 16 stay in-group)
#pragma unroll
    for (int r = 0; r < 8; ++r) {
        s[r] += __shfl_xor(s[r], 1, 32);
        s[r] += __shfl_xor(s[r], 2, 32);
        s[r] += __shfl_xor(s[r], 4, 32);
        s[r] += __shfl_xor(s[r], 8, 32);
    }

    if (ln == 0) {
        const float bias2 = b2[0];
        const int base = tile * 16 + hi * 8;    // lane0 -> rows 0..7, lane16 -> 8..15
#pragma unroll
        for (int r = 0; r < 8; ++r) {
            const int row = base + r;
            if (row < N)
                out[row] = 1.0f / (1.0f + __expf(-(s[r] + bias2)));
        }
    }
}

// ---------------------------------------------------------------------------
// Inputs (setup_inputs order): x[N*128] f32, src[N*K] int, dst (unused),
// k (unused scalar), W1[128*64] f32, b1[64] f32, W2[64] f32, b2[1] f32.
// Output: N f32. Workspace: 16KB packed W1 + padded f16 variance.
// ---------------------------------------------------------------------------
extern "C" void kernel_launch(void* const* d_in, const int* in_sizes, int n_in,
                              void* d_out, int out_size, void* d_ws, size_t ws_size,
                              hipStream_t stream) {
    const float* x   = (const float*)d_in[0];
    const int*   src = (const int*)d_in[1];
    const float* W1  = (const float*)d_in[4];
    const float* b1  = (const float*)d_in[5];
    const float* W2  = (const float*)d_in[6];
    const float* b2  = (const float*)d_in[7];
    float* out = (float*)d_out;

    const int N = in_sizes[0] / CCH;
    const int K = in_sizes[1] / N;

    _Float16* pw1 = (_Float16*)d_ws;                        // 512*16*2 = 16 KB
    _Float16* vws = (_Float16*)((char*)d_ws + 16384);       // f16 [nTiles*16,128]

    const int nTiles = (N + 15) / 16;

    pack_w1_kernel<<<1, 512, 0, stream>>>(W1, pw1);
    variance_kernel<<<(N + 7) / 8, 256, 0, stream>>>(x, src, vws, N, K, 1.0f / (float)K);
    mlp_kernel<<<(nTiles + 7) / 8, 256, 0, stream>>>(vws, pw1, b1, W2, b2, out, N, nTiles);
}
